// MultiHeadAttention_27367531610862
// MI455X (gfx1250) — matrix operand order repaired and checked by reading, not compiled, not run on hardware
//
#include <hip/hip_runtime.h>

// ---------------------------------------------------------------------------
// Multi-head attention forward for MI455X (gfx1250), wave32 + WMMA bf16.
// B=2, S=2048, D=1024, H=16, depth=64.
// d_out = [out (B*S*D f32)] ++ [attn (B*H*S*S f32)]
//
// Pipeline:
//   1. one-time f32->bf16 convert of q,k,v and wq,wk,wv,wo into workspace
//   2. Q/K/V projections: pure-bf16 WMMA GEMM, 16x64 tile per wave
//   3. logits = QK^T/8 + mask  (bf16 WMMA, f32 out into d_out attn region)
//   4. row softmax in place
//   5. P@V with V_h (256KB) resident in LDS, transposed for ds_load_b128 B-frags
//   6. output projection (bf16 WMMA, f32 out + bias)
// ---------------------------------------------------------------------------

#define B_   2
#define S_   2048
#define D_   1024
#define H_   16
#define DH_  64
#define LDSTR 2056   // padded K-stride of LDS-resident V^T (ushorts); 2056%8==0

typedef __attribute__((ext_vector_type(16))) __bf16 v16bf;
typedef __attribute__((ext_vector_type(8)))  float  v8f;

union FragBF {
    v16bf          bf;
    uint4          u4[2];
    unsigned short us[16];
};

#define WMMA_BF16(a, b, c) \
    __builtin_amdgcn_wmma_f32_16x16x32_bf16(false, (a), false, (b), (short)0, (c), false, false)

// round-to-nearest-even f32 -> bf16 bit pattern
__device__ __forceinline__ unsigned short f32_to_bf16(float f) {
    union { float f; unsigned int u; } c;
    c.f = f;
    unsigned int u = c.u + 0x7FFFu + ((c.u >> 16) & 1u);
    return (unsigned short)(u >> 16);
}

__device__ __forceinline__ void pack8(FragBF& f, int base, float4 a, float4 b) {
    f.us[base + 0] = f32_to_bf16(a.x); f.us[base + 1] = f32_to_bf16(a.y);
    f.us[base + 2] = f32_to_bf16(a.z); f.us[base + 3] = f32_to_bf16(a.w);
    f.us[base + 4] = f32_to_bf16(b.x); f.us[base + 5] = f32_to_bf16(b.y);
    f.us[base + 6] = f32_to_bf16(b.z); f.us[base + 7] = f32_to_bf16(b.w);
}

// ---- A fragment (16x32 bf16): row = lane%16; half = lane/16.
//      VGPR0-3 hold K = half*8..+7 ; VGPR4-7 hold K = 16+half*8..+7.
__device__ __forceinline__ v16bf fragA_f32(const float* row, int k0, int half) {
    const float4* p0 = (const float4*)(row + k0 + half * 8);
    const float4* p1 = (const float4*)(row + k0 + 16 + half * 8);
    FragBF f;
    pack8(f, 0, p0[0], p0[1]);
    pack8(f, 8, p1[0], p1[1]);
    return f.bf;
}

__device__ __forceinline__ v16bf fragA_bf16(const unsigned short* row, int k0, int half) {
    FragBF f;
    f.u4[0] = *(const uint4*)(row + k0 + half * 8);
    f.u4[1] = *(const uint4*)(row + k0 + 16 + half * 8);
    return f.bf;
}

// ---- B fragment (32x16 bf16): col = lane%16; VGPR0-3 hold K = half*8..+7 ;
//      VGPR4-7 hold K = 16+half*8..+7 (same K order as A).
__device__ __forceinline__ v16bf fragB_bf16(const unsigned short* row, int k0, int half) {
    FragBF f;
    f.u4[0] = *(const uint4*)(row + k0 + half * 8);
    f.u4[1] = *(const uint4*)(row + k0 + 16 + half * 8);
    return f.bf;
}

// ---------------------------------------------------------------------------
// Kernel 0: f32 -> bf16 bulk convert (n multiple of 2048)
// ---------------------------------------------------------------------------
__global__ void convert_bf16_kernel(const float* __restrict__ src,
                                    unsigned short* __restrict__ dst) {
    const size_t base = ((size_t)blockIdx.x * 256 + threadIdx.x) * 8;
    const float4 a = *(const float4*)(src + base);
    const float4 b = *(const float4*)(src + base + 4);
    FragBF f;
    pack8(f, 0, a, b);
    *(uint4*)(dst + base) = f.u4[0];
}

// ---------------------------------------------------------------------------
// Kernel 1: Y(bf16)[M x 1024] = A(bf16) @ W(bf16)^T + bias(f32)
// grid (M/16, 4), block 128: wave w handles 16x64 tile (4 accumulators)
// ---------------------------------------------------------------------------
__global__ void linear_bf16out(const unsigned short* __restrict__ A,
                               const unsigned short* __restrict__ W,
                               const float* __restrict__ bias,
                               unsigned short* __restrict__ Y) {
    const int lane = threadIdx.x & 31;
    const int wave = threadIdx.x >> 5;
    const int m16  = lane & 15;
    const int half = lane >> 4;
    const int tm0  = blockIdx.x * 16;
    const int tn   = (blockIdx.y * 4 + wave) * 64;

    const unsigned short* rowA = A + (size_t)(tm0 + m16) * D_;
    const unsigned short* rowW[4];
#pragma unroll
    for (int j = 0; j < 4; ++j)
        rowW[j] = W + (size_t)(tn + j * 16 + m16) * D_;

    v8f acc[4] = {};
    for (int k0 = 0; k0 < D_; k0 += 32) {
        const v16bf a = fragA_bf16(rowA, k0, half);
#pragma unroll
        for (int j = 0; j < 4; ++j)
            acc[j] = WMMA_BF16(a, fragB_bf16(rowW[j], k0, half), acc[j]);
    }
#pragma unroll
    for (int j = 0; j < 4; ++j) {
        const int col = tn + j * 16 + m16;
        const float bv = bias[col];
#pragma unroll
        for (int r = 0; r < 8; ++r)
            Y[(size_t)(tm0 + r + 8 * half) * D_ + col] = f32_to_bf16(acc[j][r] + bv);
    }
}

// ---------------------------------------------------------------------------
// Kernel 2: logits[b,h,m,n] = (Q_h[m,:] . K_h[n,:]) / 8 + mask[b,n] * -1e9
// grid (S/16, 8, B*H), block 128: wave covers 16x64 (4 n-tiles, A reused)
// ---------------------------------------------------------------------------
__global__ void logits_kernel(const unsigned short* __restrict__ Qp,
                              const unsigned short* __restrict__ Kp,
                              const float* __restrict__ mask,
                              float* __restrict__ attn) {
    const int lane = threadIdx.x & 31;
    const int wave = threadIdx.x >> 5;
    const int m16  = lane & 15;
    const int half = lane >> 4;
    const int b    = blockIdx.z >> 4;
    const int h    = blockIdx.z & 15;
    const int tm0  = blockIdx.x * 16;
    const int tn   = (blockIdx.y * 16 + wave * 4) * 16;  // 64 cols per wave

    const unsigned short* rowA = Qp + ((size_t)b * S_ + tm0 + m16) * D_ + h * DH_;

    v8f acc[4] = {};
#pragma unroll
    for (int k0 = 0; k0 < DH_; k0 += 32) {
        const v16bf a = fragA_bf16(rowA, k0, half);
#pragma unroll
        for (int j = 0; j < 4; ++j) {
            const unsigned short* rowB =
                Kp + ((size_t)b * S_ + tn + j * 16 + m16) * D_ + h * DH_;
            acc[j] = WMMA_BF16(a, fragB_bf16(rowB, k0, half), acc[j]);
        }
    }
    float* base = attn + (size_t)(b * H_ + h) * S_ * S_;
#pragma unroll
    for (int j = 0; j < 4; ++j) {
        const int   col = tn + j * 16 + m16;
        const float mk  = mask[b * S_ + col] * -1e9f;
#pragma unroll
        for (int r = 0; r < 8; ++r)
            base[(size_t)(tm0 + r + 8 * half) * S_ + col] = acc[j][r] * 0.125f + mk;
    }
}

// ---------------------------------------------------------------------------
// Kernel 3: in-place row softmax over 2048 elements, 256 threads/row
// ---------------------------------------------------------------------------
__global__ void softmax_kernel(float* __restrict__ attn) {
    __shared__ float red[256];
    const int t = threadIdx.x;
    float* p = attn + (size_t)blockIdx.x * S_;

    float v[8];
    float mx = -3.0e38f;
#pragma unroll
    for (int i = 0; i < 8; ++i) {
        v[i] = p[t + i * 256];
        mx = fmaxf(mx, v[i]);
    }
    red[t] = mx;
    __syncthreads();
    for (int off = 128; off > 0; off >>= 1) {
        if (t < off) red[t] = fmaxf(red[t], red[t + off]);
        __syncthreads();
    }
    mx = red[0];
    __syncthreads();

    float sum = 0.0f;
#pragma unroll
    for (int i = 0; i < 8; ++i) {
        v[i] = __expf(v[i] - mx);
        sum += v[i];
    }
    red[t] = sum;
    __syncthreads();
    for (int off = 128; off > 0; off >>= 1) {
        if (t < off) red[t] += red[t + off];
        __syncthreads();
    }
    const float inv = 1.0f / red[0];
#pragma unroll
    for (int i = 0; i < 8; ++i)
        p[t + i * 256] = v[i] * inv;
}

// ---------------------------------------------------------------------------
// Kernel 4: ctx_h = P @ V_h with V_h (2048x64 bf16 = 256KB) resident in LDS,
// stored transposed [n][k] (stride LDSTR) so each B fragment = 2x ds_load_b128.
// grid (4, 1, B*H), block 256 (8 waves); each block: 32 m-tiles; each wave:
// full 16x64 tile (4 accumulators). Dynamic LDS = 64*LDSTR*2 bytes (~257KB),
// exploiting the 320KB CDNA5 WGP LDS.
// ---------------------------------------------------------------------------
__global__ void attn_v_kernel(const float* __restrict__ attn,
                              const unsigned short* __restrict__ Vp,
                              unsigned short* __restrict__ ctx) {
    extern __shared__ unsigned short vt[];  // vt[n * LDSTR + k]

    const int lane = threadIdx.x & 31;
    const int wave = threadIdx.x >> 5;
    const int m16  = lane & 15;
    const int half = lane >> 4;
    const int b    = blockIdx.z >> 4;
    const int h    = blockIdx.z & 15;

    // ---- stage V_h transposed into LDS (one-time, amortized over 32 m-tiles)
    const unsigned short* Vh = Vp + (size_t)b * S_ * D_ + h * DH_;
    for (int idx = threadIdx.x; idx < (S_ * DH_) / 8; idx += 256) {
        const int k  = idx >> 3;
        const int cg = idx & 7;  // column group of 8
        uint4 d = *(const uint4*)(Vh + (size_t)k * D_ + cg * 8);
        const unsigned short* ds = (const unsigned short*)&d;
#pragma unroll
        for (int j = 0; j < 8; ++j)
            vt[(cg * 8 + j) * LDSTR + k] = ds[j];
    }
    __syncthreads();

    unsigned short* cbase = ctx + (size_t)b * S_ * D_ + h * DH_;

    for (int mt = wave; mt < 32; mt += 8) {
        const int tm0 = (blockIdx.x * 32 + mt) * 16;
        const float* rowA =
            attn + ((size_t)(b * H_ + h) * S_ + tm0 + m16) * S_;

        v8f acc[4] = {};
        for (int k0 = 0; k0 < S_; k0 += 32) {
            __builtin_prefetch((const void*)(rowA + k0 + 256), 0, 1);
            const v16bf a = fragA_f32(rowA, k0, half);
#pragma unroll
            for (int j = 0; j < 4; ++j) {
                FragBF fb;
                const unsigned short* src =
                    vt + (j * 16 + m16) * LDSTR + k0 + half * 8;
                fb.u4[0] = *(const uint4*)(src);
                fb.u4[1] = *(const uint4*)(src + 16);
                acc[j] = WMMA_BF16(a, fb.bf, acc[j]);
            }
        }
#pragma unroll
        for (int j = 0; j < 4; ++j)
#pragma unroll
            for (int r = 0; r < 8; ++r)
                cbase[(size_t)(tm0 + r + 8 * half) * D_ + j * 16 + m16] =
                    f32_to_bf16(acc[j][r]);
    }
}

// ---------------------------------------------------------------------------
// Kernel 5: out(f32) = ctx(bf16) @ wo(bf16)^T + bo
// ---------------------------------------------------------------------------
__global__ void linear_f32out(const unsigned short* __restrict__ A,
                              const unsigned short* __restrict__ W,
                              const float* __restrict__ bias,
                              float* __restrict__ Y) {
    const int lane = threadIdx.x & 31;
    const int wave = threadIdx.x >> 5;
    const int m16  = lane & 15;
    const int half = lane >> 4;
    const int tm0  = blockIdx.x * 16;
    const int tn   = (blockIdx.y * 4 + wave) * 64;

    const unsigned short* rowA = A + (size_t)(tm0 + m16) * D_;
    const unsigned short* rowW[4];
#pragma unroll
    for (int j = 0; j < 4; ++j)
        rowW[j] = W + (size_t)(tn + j * 16 + m16) * D_;

    v8f acc[4] = {};
    for (int k0 = 0; k0 < D_; k0 += 32) {
        const v16bf a = fragA_bf16(rowA, k0, half);
#pragma unroll
        for (int j = 0; j < 4; ++j)
            acc[j] = WMMA_BF16(a, fragB_bf16(rowW[j], k0, half), acc[j]);
    }
#pragma unroll
    for (int j = 0; j < 4; ++j) {
        const int col = tn + j * 16 + m16;
        const float bv = bias[col];
#pragma unroll
        for (int r = 0; r < 8; ++r)
            Y[(size_t)(tm0 + r + 8 * half) * D_ + col] = acc[j][r] + bv;
    }
}

// ---------------------------------------------------------------------------
extern "C" void kernel_launch(void* const* d_in, const int* in_sizes, int n_in,
                              void* d_out, int out_size, void* d_ws, size_t ws_size,
                              hipStream_t stream) {
    (void)in_sizes; (void)n_in; (void)out_size; (void)ws_size;

    const float* q    = (const float*)d_in[0];
    const float* k    = (const float*)d_in[1];
    const float* v    = (const float*)d_in[2];
    const float* mask = (const float*)d_in[3];
    const float* wq   = (const float*)d_in[4];
    const float* bq   = (const float*)d_in[5];
    const float* wk   = (const float*)d_in[6];
    const float* bk   = (const float*)d_in[7];
    const float* wv   = (const float*)d_in[8];
    const float* bv   = (const float*)d_in[9];
    const float* wo   = (const float*)d_in[10];
    const float* bo   = (const float*)d_in[11];

    float* out  = (float*)d_out;
    float* attn = out + (size_t)B_ * S_ * D_;  // second output

    const size_t NTOK = (size_t)B_ * S_ * D_;  // 4M elements
    const size_t NW   = (size_t)D_ * D_;       // 1M elements
    unsigned short* ws0 = (unsigned short*)d_ws;
    unsigned short* qb  = ws0;                 // bf16 activations
    unsigned short* kb  = qb + NTOK;
    unsigned short* vb  = kb + NTOK;
    unsigned short* wqb = vb + NTOK;           // bf16 weights
    unsigned short* wkb = wqb + NW;
    unsigned short* wvb = wkb + NW;
    unsigned short* wob = wvb + NW;
    unsigned short* Qp  = wob + NW;            // bf16 projections
    unsigned short* Kp  = Qp + NTOK;
    unsigned short* Vp  = Kp + NTOK;
    unsigned short* ctx = Vp + NTOK;           // 64 MB workspace total

    const dim3 b256(256);
    // ---- one-time f32 -> bf16 conversions
    convert_bf16_kernel<<<dim3((unsigned)(NTOK / 2048)), b256, 0, stream>>>(q, qb);
    convert_bf16_kernel<<<dim3((unsigned)(NTOK / 2048)), b256, 0, stream>>>(k, kb);
    convert_bf16_kernel<<<dim3((unsigned)(NTOK / 2048)), b256, 0, stream>>>(v, vb);
    convert_bf16_kernel<<<dim3((unsigned)(NW / 2048)), b256, 0, stream>>>(wq, wqb);
    convert_bf16_kernel<<<dim3((unsigned)(NW / 2048)), b256, 0, stream>>>(wk, wkb);
    convert_bf16_kernel<<<dim3((unsigned)(NW / 2048)), b256, 0, stream>>>(wv, wvb);
    convert_bf16_kernel<<<dim3((unsigned)(NW / 2048)), b256, 0, stream>>>(wo, wob);

    const dim3 blk(128);
    const dim3 gLin(B_ * S_ / 16, D_ / 256);   // (256, 4)

    linear_bf16out<<<gLin, blk, 0, stream>>>(qb, wqb, bq, Qp);
    linear_bf16out<<<gLin, blk, 0, stream>>>(kb, wkb, bk, Kp);
    linear_bf16out<<<gLin, blk, 0, stream>>>(vb, wvb, bv, Vp);

    logits_kernel<<<dim3(S_ / 16, S_ / 256, B_ * H_), blk, 0, stream>>>(Qp, Kp, mask, attn);

    softmax_kernel<<<dim3(B_ * H_ * S_), b256, 0, stream>>>(attn);

    const size_t ldsBytes = (size_t)DH_ * LDSTR * sizeof(unsigned short);  // ~257 KB
    attn_v_kernel<<<dim3(4, 1, B_ * H_), b256, ldsBytes, stream>>>(attn, Vp, ctx);

    linear_f32out<<<gLin, blk, 0, stream>>>(ctx, wob, bo, out);
}
